// EmbbedingGNN_42726334660669
// MI455X (gfx1250) — matrix operand order repaired and checked by reading, not compiled, not run on hardware
//
#include <hip/hip_runtime.h>
#include <cstdint>
#include <cstddef>

typedef _Float16 half_t;
typedef __attribute__((ext_vector_type(16))) _Float16 v16h;
typedef __attribute__((ext_vector_type(8)))  float    v8f;

#define DD 128   // hidden dim
#define NH 8     // heads

// explicit global-address-space 64-bit load (forces global_load_b64, not flat)
typedef const unsigned long long __attribute__((address_space(1)))* gu64p;
static __device__ __forceinline__ unsigned long long ld_g64(const float* p) {
  return *(gu64p)(uintptr_t)p;
}

// ---------------------------------------------------------------------------
// helpers
// ---------------------------------------------------------------------------
static __device__ __forceinline__ void atomicMaxF(float* addr, float val) {
  if (val >= 0.0f) atomicMax((int*)addr, __float_as_int(val));
  else             atomicMin((unsigned int*)addr, __float_as_uint(val));
}

// ---------------------------------------------------------------------------
// one-shot weight transpose + f32->f16: WT[n*K + k] = (half)W[k*128 + n]
// ---------------------------------------------------------------------------
__global__ void k_wtrans(const float* __restrict__ W, half_t* __restrict__ WT, int K)
{
  long long t = (long long)blockIdx.x * blockDim.x + threadIdx.x;
  if (t >= (long long)K * DD) return;
  int n = (int)(t / K);
  int k = (int)(t - (long long)n * K);
  WT[t] = (half_t)W[(size_t)k * DD + n];
}

// ---------------------------------------------------------------------------
// WMMA GEMM: out[M,128] = concat(A0..A5)[M,K] @ W[K,128] (+bias) (+relu)
// W pre-transposed to f16 WT[128][K]. Block = 256 thr = 8 wave32.
// Block tile = 32 rows x 128 cols; wave w -> cols [16w,16w+16), 2 row tiles.
//
// A fragment LDS layout: staged pre-permuted so each lane reads its whole
// v16h fragment as 32 contiguous bytes (2x ds_load_b128).
// Inverse map for element (m,k) of the 16x32 f16 A tile (contiguous k order, same as B):
//   lane = m + ((k>=16)?16:0)
//   vgpr = (k&15)>>1,  half = k&1
// A b64 load at even k fills halves (k,k+1) of one lane's vgpr -> one
// packed b32 LDS store per staging thread. 1-deep software pipeline on the
// staging loads (prefetch next k-step while WMMA runs).
// ---------------------------------------------------------------------------
__global__ void __launch_bounds__(256)
k_wmma_gemm(const float* __restrict__ A0, const float* __restrict__ A1,
            const float* __restrict__ A2, const float* __restrict__ A3,
            const float* __restrict__ A4, const float* __restrict__ A5,
            const half_t* __restrict__ WT, const float* __restrict__ bias,
            float* __restrict__ out, int M, int K, int relu)
{
  __shared__ __align__(32) uint32_t Afrag[2][32][8];   // [row-tile][lane][vgpr]
  const float* segs[6] = {A0, A1, A2, A3, A4, A5};

  const int tid  = threadIdx.x;
  const int wave = tid >> 5;
  const int lane = tid & 31;
  const int brow = blockIdx.x * 32;

  const int n     = (wave << 4) + (lane & 15);       // output column
  const int bkoff = (lane >= 16) ? 16 : 0;           // B fragment k offset
  const half_t* wt_n = WT + (size_t)n * K;

  // loop-invariant staging descriptors: element0 = row m, element1 = row m+16
  const int m  = tid >> 4;                            // 0..15
  const int p  = tid & 15;                            // k pair, k = 2p
  const int gr0 = brow + m;
  const int gr1 = gr0 + 16;
  const bool z0 = (gr0 >= M), z1 = (gr1 >= M);
  const size_t ro0 = (size_t)(z0 ? (M - 1) : gr0) * DD + 2 * p;
  const size_t ro1 = (size_t)(z1 ? (M - 1) : gr1) * DD + 2 * p;
  const int ln = m + ((p >= 8) ? 16 : 0);             // k>=16
  const int vg = (p & 7);                             // (k&15)>>1
  uint32_t* dst0 = &Afrag[0][ln][vg];
  uint32_t* dst1 = &Afrag[1][ln][vg];

  v8f acc0 = {}, acc1 = {};

  // prologue: prefetch k0 = 0
  unsigned long long c0 = ld_g64(segs[0] + ro0);
  unsigned long long c1 = ld_g64(segs[0] + ro1);

  for (int k0 = 0; k0 < K; k0 += 32) {
    // ---- convert + store current staged pair ----
    {
      float x0 = __builtin_bit_cast(float, (uint32_t)c0);
      float y0 = __builtin_bit_cast(float, (uint32_t)(c0 >> 32));
      float x1 = __builtin_bit_cast(float, (uint32_t)c1);
      float y1 = __builtin_bit_cast(float, (uint32_t)(c1 >> 32));
      if (z0) { x0 = 0.f; y0 = 0.f; }
      if (z1) { x1 = 0.f; y1 = 0.f; }
      half_t lo = (half_t)x0, hi = (half_t)y0;
      *dst0 = (uint32_t)__builtin_bit_cast(uint16_t, lo) |
              ((uint32_t)__builtin_bit_cast(uint16_t, hi) << 16);
      lo = (half_t)x1; hi = (half_t)y1;
      *dst1 = (uint32_t)__builtin_bit_cast(uint16_t, lo) |
              ((uint32_t)__builtin_bit_cast(uint16_t, hi) << 16);
    }
    __syncthreads();

    // ---- prefetch next k-step (overlaps with frag reads + WMMA) ----
    int kn = k0 + 32;
    if (kn < K) {
      const float* base = segs[kn >> 7];
      int kc = kn & 127;
      c0 = ld_g64(base + ro0 + kc);
      c1 = ld_g64(base + ro1 + kc);
    }

    // ---- fragments: A = 2x b128 from LDS, B = 2x b128 from L2 ----
    v16h a0 = *(const v16h*)&Afrag[0][lane][0];
    v16h a1 = *(const v16h*)&Afrag[1][lane][0];
    v16h b  = *(const v16h*)(wt_n + k0 + bkoff);

    acc0 = __builtin_amdgcn_wmma_f32_16x16x32_f16(false, a0, false, b, (short)0, acc0, false, false);
    acc1 = __builtin_amdgcn_wmma_f32_16x16x32_f16(false, a1, false, b, (short)0, acc1, false, false);
    __syncthreads();
  }

  // ---- epilogue: lane L, vgpr v -> row v + (L>=16?8:0), col n ----
  const int mhi = (lane >= 16) ? 8 : 0;
  float bv = bias ? bias[n] : 0.0f;
#pragma unroll
  for (int v = 0; v < 8; ++v) {
    int go0 = brow + v + mhi;
    int go1 = go0 + 16;
    if (go0 < M) {
      float val = acc0[v] + bv;
      if (relu) val = fmaxf(val, 0.0f);
      out[(size_t)go0 * DD + n] = val;
    }
    if (go1 < M) {
      float val = acc1[v] + bv;
      if (relu) val = fmaxf(val, 0.0f);
      out[(size_t)go1 * DD + n] = val;
    }
  }
}

// ---------------------------------------------------------------------------
// small dense kernels
// ---------------------------------------------------------------------------
__global__ void k_uplift(const float* __restrict__ X, const float* __restrict__ W,
                         const float* __restrict__ b, float* __restrict__ out,
                         long long N, int F)
{
  long long t = (long long)blockIdx.x * blockDim.x + threadIdx.x;
  if (t >= N * DD) return;
  long long n = t >> 7; int c = (int)(t & 127);
  float acc = b[c];
  for (int f = 0; f < F; ++f) acc += X[n * F + f] * W[(size_t)f * DD + c];
  out[t] = acc;
}

// wfold[d,h] = sum_c W[d, h*16+c] * att[h*16+c]
__global__ void k_fold_att(const float* __restrict__ W, const float* __restrict__ att,
                           float* __restrict__ wfold)
{
  int t = blockIdx.x * blockDim.x + threadIdx.x;
  if (t >= DD * NH) return;
  int d = t >> 3, h = t & 7;
  float s = 0.f;
  for (int c = 0; c < 16; ++c) s += W[(size_t)d * DD + h * 16 + c] * att[h * 16 + c];
  wfold[d * NH + h] = s;
}

// wchain[f,h] = Wup[f,:] . wfold[:,h] ; bchain[h] = bup . wfold[:,h]
__global__ void k_chain_fold(const float* __restrict__ Wup, const float* __restrict__ bup,
                             const float* __restrict__ wfold, float* __restrict__ wchain,
                             float* __restrict__ bchain, int F)
{
  int t = blockIdx.x * blockDim.x + threadIdx.x;
  if (t >= (F + 1) * NH) return;
  int row = t >> 3, h = t & 7;
  const float* src = (row < F) ? (Wup + (size_t)row * DD) : bup;
  float s = 0.f;
  for (int d = 0; d < DD; ++d) s += src[d] * wfold[d * NH + h];
  if (row < F) wchain[row * NH + h] = s; else bchain[h] = s;
}

// ae[e,h] = attr[e,:F] @ wchain[:,h] + bchain[h]
__global__ void k_edge_logits(const float* __restrict__ attr, const float* __restrict__ wchain,
                              const float* __restrict__ bchain, float* __restrict__ ae,
                              long long E, int F)
{
  long long t = (long long)blockIdx.x * blockDim.x + threadIdx.x;
  if (t >= E * NH) return;
  long long e = t >> 3; int h = (int)(t & 7);
  float s = bchain[h];
  for (int f = 0; f < F; ++f) s += attr[e * F + f] * wchain[f * NH + h];
  ae[t] = s;
}

// as[n,h] = sum_c hs[n, h*16+c] * att[h*16+c]
__global__ void k_att_node(const float* __restrict__ hs, const float* __restrict__ att,
                           float* __restrict__ as_, long long N)
{
  long long t = (long long)blockIdx.x * blockDim.x + threadIdx.x;
  if (t >= N * NH) return;
  long long n = t >> 3; int h = (int)(t & 7);
  float s = 0.f;
  for (int c = 0; c < 16; ++c) s += hs[n * DD + h * 16 + c] * att[h * 16 + c];
  as_[t] = s;
}

// ad[n,h] = x[n,:] . wfold[:,h]
__global__ void k_skinny8(const float* __restrict__ x, const float* __restrict__ wfold,
                          float* __restrict__ ad, long long N)
{
  long long t = (long long)blockIdx.x * blockDim.x + threadIdx.x;
  if (t >= N * NH) return;
  long long n = t >> 3; int h = (int)(t & 7);
  float s = 0.f;
  for (int d = 0; d < DD; ++d) s += x[n * DD + d] * wfold[d * NH + h];
  ad[t] = s;
}

__global__ void k_fill(float* __restrict__ p, float v, long long n)
{
  long long t = (long long)blockIdx.x * blockDim.x + threadIdx.x;
  if (t < n) p[t] = v;
}

__global__ void k_fill_bias(float* __restrict__ out, const float* __restrict__ bias,
                            long long N)
{
  long long t = (long long)blockIdx.x * blockDim.x + threadIdx.x;
  if (t < N * DD) out[t] = bias[t & 127];
}

// pass 1: raw logits + leaky relu + segment max (atomic)
__global__ void k_edge_max(const int* __restrict__ ei, long long E,
                           const float* __restrict__ as_, const float* __restrict__ ad_,
                           const float* __restrict__ ae, float* __restrict__ abuf,
                           float* __restrict__ mbuf)
{
  long long t = (long long)blockIdx.x * blockDim.x + threadIdx.x;
  if (t >= E * NH) return;
  long long e = t >> 3; int h = (int)(t & 7);
  int s = ei[e], d = ei[E + e];
  float a = as_[(size_t)s * NH + h] + ad_[(size_t)d * NH + h];
  if (ae) a += ae[t];
  a = (a > 0.f) ? a : 0.2f * a;   // leaky_relu(0.2)
  abuf[t] = a;
  atomicMaxF(&mbuf[(size_t)d * NH + h], a);
}

// pass 2: e = exp(a - m[dst]); segment sum
__global__ void k_edge_exp(const int* __restrict__ ei, long long E,
                           float* __restrict__ abuf, const float* __restrict__ mbuf,
                           float* __restrict__ sbuf)
{
  long long t = (long long)blockIdx.x * blockDim.x + threadIdx.x;
  if (t >= E * NH) return;
  long long e = t >> 3; int h = (int)(t & 7);
  int d = ei[E + e];
  float mv = mbuf[(size_t)d * NH + h];
  if (!(mv > -1e37f)) mv = 0.f;
  float ev = __expf(abuf[t] - mv);
  abuf[t] = ev;
  atomicAdd(&sbuf[(size_t)d * NH + h], ev);
}

// pass 3: out[dst,c] += hs[src,c] * alpha(e, h(c))
__global__ void k_edge_agg(const int* __restrict__ ei, long long E,
                           const float* __restrict__ abuf, const float* __restrict__ sbuf,
                           const float* __restrict__ hs, float* __restrict__ out)
{
  long long t = (long long)blockIdx.x * blockDim.x + threadIdx.x;
  if (t >= E * DD) return;
  long long e = t >> 7; int c = (int)(t & 127); int h = c >> 4;
  int s = ei[e], d = ei[E + e];
  float alpha = abuf[e * NH + h] / (sbuf[(size_t)d * NH + h] + 1e-16f);
  atomicAdd(&out[(size_t)d * DD + c], hs[(size_t)s * DD + c] * alpha);
}

// y = LN(t2 + t3) * g + b  (one 128-thread block per row)
__global__ void __launch_bounds__(128)
k_add_ln(const float* __restrict__ t2, const float* __restrict__ t3,
         const float* __restrict__ g, const float* __restrict__ b,
         float* __restrict__ out, long long N)
{
  __shared__ float red[128];
  long long row = blockIdx.x;
  int c = threadIdx.x;
  float v = t2[row * DD + c] + t3[row * DD + c];
  red[c] = v; __syncthreads();
  for (int s = 64; s > 0; s >>= 1) { if (c < s) red[c] += red[c + s]; __syncthreads(); }
  float mu = red[0] * (1.0f / DD);
  __syncthreads();
  float dv = v - mu;
  red[c] = dv * dv; __syncthreads();
  for (int s = 64; s > 0; s >>= 1) { if (c < s) red[c] += red[c + s]; __syncthreads(); }
  float var = red[0] * (1.0f / DD);
  out[row * DD + c] = dv * rsqrtf(var + 1e-5f) * g[c] + b[c];
}

// ---------------------------------------------------------------------------
// host side
// ---------------------------------------------------------------------------
struct GatP { const float *W_dst, *W_edge, *W_src, *att_dst, *att_edge, *att_src, *bias; };
struct MlpP { const float *fc1W, *fc1b, *fc2W, *fc2b, *ln_b, *ln_g, *resW, *resb; };
struct LinP { const float *W = nullptr, *b = nullptr; };

extern "C" void kernel_launch(void* const* d_in, const int* in_sizes, int n_in,
                              void* d_out, int out_size, void* d_ws, size_t ws_size,
                              hipStream_t stream) {
  (void)in_sizes; (void)n_in; (void)out_size; (void)ws_size;
  const long long N_OP = 100000, N_IT = 30000, N_MAT = 10000, N_RES = 5000;
  const long long E_NFM = 300000, E_NFR = 500000, E_OPIT = 200000;
  const long long E_PAR = 30000, E_CHI = 30000, E_OPAS = 100000;
  const long long E_PRE = 200000, E_SUC = 200000;

  const float* in_op    = (const float*)d_in[0];
  const float* in_it    = (const float*)d_in[1];
  const float* in_mat   = (const float*)d_in[2];
  const float* in_res   = (const float*)d_in[3];
  const float* nfm_attr = (const float*)d_in[4];
  const float* nfr_attr = (const float*)d_in[5];
  const float* rnfm_attr= (const float*)d_in[6];
  const float* rnfr_attr= (const float*)d_in[7];
  const float* P        = (const float*)d_in[8];
  const int* ei_nfm  = (const int*)d_in[9];
  const int* ei_nfr  = (const int*)d_in[10];
  const int* ei_opit = (const int*)d_in[11];
  const int* ei_par  = (const int*)d_in[12];
  const int* ei_chi  = (const int*)d_in[13];
  const int* ei_opas = (const int*)d_in[14];
  const int* ei_pre  = (const int*)d_in[15];
  const int* ei_suc  = (const int*)d_in[16];
  const int* ei_rnfr = (const int*)d_in[17];
  const int* ei_rnfm = (const int*)d_in[18];

  // ---- params walk (jax pytree order: dict keys sorted, lists in order) ----
  size_t off = 0;
  auto takef = [&](size_t n) { const float* p = P + off; off += n; return p; };
  // sorted edge types: children, item_for_op, mat_for_op, ops_for_item,
  //                    ops_for_mat, ops_for_res, parents, preds, res_for_op, succs
  const bool eflag[10] = {false, false, true, false, true, true, false, false, true, false};
  GatP gat[2][10];
  for (int l = 0; l < 2; ++l)
    for (int t = 0; t < 10; ++t) {
      GatP g;
      g.W_dst   = takef(16384);
      g.W_edge  = eflag[t] ? takef(16384) : nullptr;
      g.W_src   = takef(16384);
      g.att_dst = takef(128);
      g.att_edge= eflag[t] ? takef(128) : nullptr;
      g.att_src = takef(128);
      g.bias    = takef(128);
      gat[l][t] = g;
    }
  auto takeMlp = [&](int K) {
    MlpP m;
    m.fc1W = takef((size_t)K * 128); m.fc1b = takef(128);
    m.fc2W = takef(16384);           m.fc2b = takef(128);
    m.ln_b = takef(128);             m.ln_g = takef(128);
    m.resW = takef((size_t)K * 128); m.resb = takef(128);
    return m;
  };
  MlpP mlp_it = takeMlp(512), mlp_mat = takeMlp(256), mlp_op = takeMlp(768), mlp_res = takeMlp(256);
  auto takeLin = [&](int F) { LinP L; L.W = takef((size_t)F * 128); L.b = takef(128); return L; };
  LinP up_it = takeLin(14), up_mat = takeLin(10), up_nfm = takeLin(8), up_nfr = takeLin(9);
  LinP up_op = takeLin(16), up_res = takeLin(12), up_rnfm = takeLin(8), up_rnfr = takeLin(9);

  // ---- workspace bump allocator ----
  char* ws = (char*)d_ws; size_t wo = 0;
  auto alloc = [&](size_t nfloats) { float* p = (float*)(ws + wo); wo += nfloats * sizeof(float); return p; };
  float* X_OP  = alloc(N_OP * 128);  float* X_IT  = alloc(N_IT * 128);
  float* X_MAT = alloc(N_MAT * 128); float* X_RES = alloc(N_RES * 128);
  float* HS    = alloc(N_OP * 128);
  float* G_OPS_MAT = alloc(N_MAT * 128); float* G_OPS_RES = alloc(N_RES * 128);
  float* G_OPS_IT  = alloc(N_IT * 128);  float* G_PAR = alloc(N_IT * 128);
  float* G_CHI = alloc(N_IT * 128);      float* G_IT_OP = alloc(N_OP * 128);
  float* G_PRE = alloc(N_OP * 128);      float* G_SUC = alloc(N_OP * 128);
  float* G_RES_OP = alloc(N_OP * 128);   float* G_MAT_OP = alloc(N_OP * 128);
  float* AS_ = alloc(N_OP * 8); float* AD_ = alloc(N_OP * 8);
  float* AE_ = alloc(E_NFR * 8); float* ABUF = alloc(E_NFR * 8);
  float* MBUF = alloc(N_OP * 8); float* SBUF = alloc(N_OP * 8);
  float* WFOLD = alloc(1024); float* WFOLD_E = alloc(1024);
  float* WCHAIN = alloc(256); float* BCHAIN = alloc(64);
  half_t* WT = (half_t*)alloc(768 * 128 / 2);   // 768*128 halfs
  float* T1 = alloc(N_OP * 128); float* T2 = alloc(N_OP * 128); float* T3 = alloc(N_OP * 128);

  auto cdiv = [](long long a, long long b) { return (unsigned)((a + b - 1) / b); };
  auto gemm = [&](const float* a0, const float* a1, const float* a2, const float* a3,
                  const float* a4, const float* a5, const float* W, const float* bias,
                  float* out, long long M, int K, int relu) {
    k_wtrans<<<cdiv((long long)K * 128, 256), 256, 0, stream>>>(W, WT, K);
    k_wmma_gemm<<<cdiv(M, 32), 256, 0, stream>>>(a0, a1, a2, a3, a4, a5, WT, bias, out, (int)M, K, relu);
  };

  // ---- input uplifts (shared across layers) ----
  k_uplift<<<cdiv(N_OP * 128, 256), 256, 0, stream>>>(in_op,  up_op.W,  up_op.b,  X_OP,  N_OP, 16);
  k_uplift<<<cdiv(N_IT * 128, 256), 256, 0, stream>>>(in_it,  up_it.W,  up_it.b,  X_IT,  N_IT, 14);
  k_uplift<<<cdiv(N_MAT * 128, 256), 256, 0, stream>>>(in_mat, up_mat.W, up_mat.b, X_MAT, N_MAT, 10);
  k_uplift<<<cdiv(N_RES * 128, 256), 256, 0, stream>>>(in_res, up_res.W, up_res.b, X_RES, N_RES, 12);

  auto run_gat = [&](const float* xs, long long Ns, const float* xd, long long Nd,
                     const int* ei, long long E, const float* attr, int F, LinP up,
                     const GatP& g, float* out) {
    gemm(xs, 0, 0, 0, 0, 0, g.W_src, nullptr, HS, Ns, 128, 0);
    k_att_node<<<cdiv(Ns * 8, 256), 256, 0, stream>>>(HS, g.att_src, AS_, Ns);
    k_fold_att<<<4, 256, 0, stream>>>(g.W_dst, g.att_dst, WFOLD);
    k_skinny8<<<cdiv(Nd * 8, 256), 256, 0, stream>>>(xd, WFOLD, AD_, Nd);
    const float* aePtr = nullptr;
    if (attr) {
      k_fold_att<<<4, 256, 0, stream>>>(g.W_edge, g.att_edge, WFOLD_E);
      k_chain_fold<<<1, 128, 0, stream>>>(up.W, up.b, WFOLD_E, WCHAIN, BCHAIN, F);
      k_edge_logits<<<cdiv(E * 8, 256), 256, 0, stream>>>(attr, WCHAIN, BCHAIN, AE_, E, F);
      aePtr = AE_;
    }
    k_fill<<<cdiv(Nd * 8, 256), 256, 0, stream>>>(MBUF, -__builtin_inff(), Nd * 8);
    k_fill<<<cdiv(Nd * 8, 256), 256, 0, stream>>>(SBUF, 0.0f, Nd * 8);
    k_fill_bias<<<cdiv(Nd * 128, 256), 256, 0, stream>>>(out, g.bias, Nd);
    k_edge_max<<<cdiv(E * 8, 256), 256, 0, stream>>>(ei, E, AS_, AD_, aePtr, ABUF, MBUF);
    k_edge_exp<<<cdiv(E * 8, 256), 256, 0, stream>>>(ei, E, ABUF, MBUF, SBUF);
    k_edge_agg<<<cdiv(E * 128, 256), 256, 0, stream>>>(ei, E, ABUF, SBUF, HS, out);
  };

  auto run_mlp = [&](const float* s0, const float* s1, const float* s2, const float* s3,
                     const float* s4, const float* s5, int K, long long N, const MlpP& m, float* x) {
    gemm(s0, s1, s2, s3, s4, s5, m.fc1W, m.fc1b, T1, N, K, 1);
    gemm(s0, s1, s2, s3, s4, s5, m.resW, m.resb, T3, N, K, 0);
    gemm(T1, 0, 0, 0, 0, 0, m.fc2W, m.fc2b, T2, N, 128, 0);
    k_add_ln<<<(unsigned)N, 128, 0, stream>>>(T2, T3, m.ln_g, m.ln_b, x, N);
  };

  for (int l = 0; l < 2; ++l) {
    const GatP* G = gat[l];
    // sorted-order indices: 0 children,1 item_for_op,2 mat_for_op,3 ops_for_item,
    // 4 ops_for_mat,5 ops_for_res,6 parents,7 preds,8 res_for_op,9 succs
    run_gat(X_OP, N_OP, X_MAT, N_MAT, ei_nfm, E_NFM, nfm_attr, 8,  up_nfm,  G[4], G_OPS_MAT);
    run_gat(X_OP, N_OP, X_RES, N_RES, ei_nfr, E_NFR, nfr_attr, 9,  up_nfr,  G[5], G_OPS_RES);
    run_gat(X_OP, N_OP, X_IT,  N_IT,  ei_opit, E_OPIT, nullptr, 0, LinP{},  G[3], G_OPS_IT);
    run_gat(X_IT, N_IT, X_IT,  N_IT,  ei_par,  E_PAR,  nullptr, 0, LinP{},  G[6], G_PAR);
    run_gat(X_IT, N_IT, X_IT,  N_IT,  ei_chi,  E_CHI,  nullptr, 0, LinP{},  G[0], G_CHI);
    run_gat(X_IT, N_IT, X_OP,  N_OP,  ei_opas, E_OPAS, nullptr, 0, LinP{},  G[1], G_IT_OP);
    run_gat(X_OP, N_OP, X_OP,  N_OP,  ei_pre,  E_PRE,  nullptr, 0, LinP{},  G[7], G_PRE);
    run_gat(X_OP, N_OP, X_OP,  N_OP,  ei_suc,  E_SUC,  nullptr, 0, LinP{},  G[9], G_SUC);
    run_gat(X_RES, N_RES, X_OP, N_OP, ei_rnfr, E_NFR, rnfr_attr, 9, up_rnfr, G[8], G_RES_OP);
    run_gat(X_MAT, N_MAT, X_OP, N_OP, ei_rnfm, E_NFM, rnfm_attr, 8, up_rnfm, G[2], G_MAT_OP);

    run_mlp(X_MAT, G_OPS_MAT, 0, 0, 0, 0, 256, N_MAT, mlp_mat, X_MAT);
    run_mlp(X_RES, G_OPS_RES, 0, 0, 0, 0, 256, N_RES, mlp_res, X_RES);
    run_mlp(X_IT, G_OPS_IT, G_PAR, G_CHI, 0, 0, 512, N_IT, mlp_it, X_IT);
    run_mlp(X_OP, G_PRE, G_SUC, G_IT_OP, G_RES_OP, G_MAT_OP, 768, N_OP, mlp_op, X_OP);
  }

  // outputs: (x_mat, x_res, x_it, x_op) concatenated
  float* out = (float*)d_out;
  (void)hipMemcpyAsync(out,                                X_MAT, N_MAT * 128 * 4, hipMemcpyDeviceToDevice, stream);
  (void)hipMemcpyAsync(out + N_MAT * 128,                  X_RES, N_RES * 128 * 4, hipMemcpyDeviceToDevice, stream);
  (void)hipMemcpyAsync(out + (N_MAT + N_RES) * 128,        X_IT,  N_IT  * 128 * 4, hipMemcpyDeviceToDevice, stream);
  (void)hipMemcpyAsync(out + (N_MAT + N_RES + N_IT) * 128, X_OP,  N_OP  * 128 * 4, hipMemcpyDeviceToDevice, stream);
}
